// SpatialGraphConv_87162066305609
// MI455X (gfx1250) — compile-verified, matrix operand order checked by reading, not run on hardware
//
#include <hip/hip_runtime.h>

typedef __attribute__((ext_vector_type(16))) _Float16 v16h;
typedef __attribute__((ext_vector_type(8)))  _Float16 v8h;
typedef __attribute__((ext_vector_type(8)))  float    v8f;

#define NBT   8192      // B*T = 32*256
#define NV    86
#define VP    96        // padded V (6 x 16)
#define CIN   64
#define COUT  128

union Frag { v16h v; v8h h[2]; };

// Load a 16x32 f16 WMMA A/B fragment from LDS (row-major, k-contiguous).
// p points at &buf[row * ld + k0 + khalf]; element i of the fragment is
// buf[row][k0 + (i<8?0:16) + khalf + (i&7)]  ->  two contiguous 16B chunks.
static __device__ inline v16h ld_frag(const _Float16* p) {
  Frag f;
  f.h[0] = *(const v8h*)(p);
  f.h[1] = *(const v8h*)(p + 16);
  return f.v;
}

// ---------------------------------------------------------------------------
// Kernel 1: normalized adjacency (padded to 96x96, f16) + zero stat buffers.
// ---------------------------------------------------------------------------
__global__ void gcn_prep_kernel(const float* __restrict__ adj,
                                _Float16* __restrict__ nA,
                                float* __restrict__ stats /*256 floats*/) {
  __shared__ float dis[VP];
  const int t = threadIdx.x;   // 128 threads
  if (t < VP) {
    float d = 0.f;
    if (t < NV) {
      float s = 1.0f;                       // +I on the diagonal
      for (int v = 0; v < NV; ++v) s += adj[t * NV + v];
      d = rsqrtf(s);
    }
    dis[t] = d;
  }
  stats[t] = 0.f;          // gSum
  stats[t + 128] = 0.f;    // gSumSq
  __syncthreads();
  for (int e = t; e < VP * VP; e += 128) {
    int u = e / VP, v = e % VP;
    float val = 0.f;
    if (u < NV && v < NV) {
      float a = adj[u * NV + v] + (u == v ? 1.f : 0.f);
      val = dis[u] * a * dis[v];
    }
    nA[e] = (_Float16)val;
  }
}

// ---------------------------------------------------------------------------
// Kernel 2/4: fused  normA @ X @ W  per (b,t).  MODE 0: accumulate per-channel
// sum / sumsq.  MODE 1: apply BN scale/shift + ReLU and write output.
// ---------------------------------------------------------------------------
template <int MODE>
__global__ __launch_bounds__(256) void gcn_gemm_kernel(
    const float* __restrict__ x, const _Float16* __restrict__ nA,
    const float* __restrict__ w,
    float* __restrict__ gSum, float* __restrict__ gSumSq,
    const float* __restrict__ gScale, const float* __restrict__ gShift,
    float* __restrict__ out) {
  __shared__ __align__(16) _Float16 sA[VP * VP];      // normA  [u][v]
  __shared__ __align__(16) _Float16 sXt[CIN * VP];    // X^T    [c][v]
  __shared__ __align__(16) _Float16 sY1[VP * CIN];    // Y1     [u][c]
  __shared__ __align__(16) _Float16 sWt[COUT * CIN];  // W^T    [d][c]
  __shared__ float sSum[COUT];
  __shared__ float sSq[COUT];

  const int t = threadIdx.x;
  const int lane = t & 31;
  const int wv = t >> 5;          // wave 0..7
  const int bt = blockIdx.x;

  if (MODE == 0 && t < COUT) { sSum[t] = 0.f; sSq[t] = 0.f; }

  // --- stage operands into LDS (f16) ---
  {
    const unsigned int* src = (const unsigned int*)nA;   // 96*96 f16 = 4608 dwords
    unsigned int* dst = (unsigned int*)sA;
    for (int e = t; e < VP * VP / 2; e += 256) dst[e] = src[e];
  }
  {
    const float* xb = x + (size_t)bt * (NV * CIN);
    for (int e = t; e < VP * CIN; e += 256) {
      int v = e >> 6, c = e & 63;                         // coalesced read
      float val = (v < NV) ? xb[v * CIN + c] : 0.f;       // zero-pad rows
      sXt[c * VP + v] = (_Float16)val;                    // k-contiguous store
    }
  }
  {
    for (int e = t; e < CIN * COUT; e += 256) {
      int c = e >> 7, d = e & 127;                        // coalesced read
      sWt[d * CIN + c] = (_Float16)w[c * COUT + d];
    }
  }
  __syncthreads();

  const int m = lane & 15;                 // row/col within tile
  const int khalf = (lane >> 4) << 3;      // 0 or 8
  const int rbase_half = (lane >> 4) << 3; // 0 or 8 (C/D row offset)

  // --- GEMM1: Y1[96x64] = normA[96x96] @ X[96x64]  (24 tiles / 8 waves) ---
  for (int j = 0; j < 3; ++j) {
    const int tile = wv * 3 + j;
    const int mi = tile >> 2, ni = tile & 3;
    v8f acc = {};
    for (int ks = 0; ks < 3; ++ks) {
      const int k0 = ks * 32;
      v16h a = ld_frag(&sA[(mi * 16 + m) * VP + k0 + khalf]);
      v16h b = ld_frag(&sXt[(ni * 16 + m) * VP + k0 + khalf]);
      acc = __builtin_amdgcn_wmma_f32_16x16x32_f16(false, a, false, b,
                                                   (short)0, acc, false, false);
    }
    const int rbase = mi * 16 + rbase_half;
    const int col = ni * 16 + m;
    for (int i = 0; i < 8; ++i)
      sY1[(rbase + i) * CIN + col] = (_Float16)acc[i];
  }
  __syncthreads();

  // --- GEMM2: Y2[96x128] = Y1[96x64] @ W[64x128]  (48 tiles / 8 waves) ---
  for (int j = 0; j < 6; ++j) {
    const int tile = wv * 6 + j;
    const int mi = tile >> 3, ni = tile & 7;
    v8f acc = {};
    for (int ks = 0; ks < 2; ++ks) {
      const int k0 = ks * 32;
      v16h a = ld_frag(&sY1[(mi * 16 + m) * CIN + k0 + khalf]);
      v16h b = ld_frag(&sWt[(ni * 16 + m) * CIN + k0 + khalf]);
      acc = __builtin_amdgcn_wmma_f32_16x16x32_f16(false, a, false, b,
                                                   (short)0, acc, false, false);
    }
    const int c = ni * 16 + m;
    const int rbase = mi * 16 + rbase_half;
    if (MODE == 0) {
      // padded rows are exactly zero -> contribute nothing to sums
      float s = 0.f, q = 0.f;
      for (int i = 0; i < 8; ++i) { float v2 = acc[i]; s += v2; q += v2 * v2; }
      atomicAdd(&sSum[c], s);
      atomicAdd(&sSq[c], q);
    } else {
      const float sc = gScale[c], sh = gShift[c];
      float* ob = out + (size_t)bt * (NV * COUT);
      for (int i = 0; i < 8; ++i) {
        const int r = rbase + i;
        if (r < NV) {
          float v2 = acc[i] * sc + sh;
          ob[r * COUT + c] = v2 > 0.f ? v2 : 0.f;
        }
      }
    }
  }

  if (MODE == 0) {
    __syncthreads();
    if (t < COUT) {
      unsafeAtomicAdd(&gSum[t], sSum[t]);
      unsafeAtomicAdd(&gSumSq[t], sSq[t]);
    }
  }
}

// ---------------------------------------------------------------------------
// Kernel 3: fold BN stats + gamma/beta into per-channel scale/shift.
// ---------------------------------------------------------------------------
__global__ void gcn_finalize_kernel(const float* __restrict__ gSum,
                                    const float* __restrict__ gSumSq,
                                    const float* __restrict__ gamma,
                                    const float* __restrict__ beta,
                                    float* __restrict__ gScale,
                                    float* __restrict__ gShift) {
  const int c = threadIdx.x;  // 128 threads
  const float n = 32.f * 256.f * 86.f;
  const float mean = gSum[c] / n;
  const float var = gSumSq[c] / n - mean * mean;
  const float sc = gamma[c] * rsqrtf(var + 1e-5f);
  gScale[c] = sc;
  gShift[c] = beta[c] - mean * sc;
}

extern "C" void kernel_launch(void* const* d_in, const int* in_sizes, int n_in,
                              void* d_out, int out_size, void* d_ws, size_t ws_size,
                              hipStream_t stream) {
  const float* x     = (const float*)d_in[0];
  const float* adj   = (const float*)d_in[1];
  const float* w     = (const float*)d_in[2];
  const float* gamma = (const float*)d_in[3];
  const float* beta  = (const float*)d_in[4];
  float* out = (float*)d_out;

  // workspace layout: [96*96 f16 normA][128 sum][128 sumsq][128 scale][128 shift]
  _Float16* nA  = (_Float16*)d_ws;
  float* stats  = (float*)((char*)d_ws + VP * VP * sizeof(_Float16));
  float* gSum   = stats;
  float* gSumSq = stats + 128;
  float* gScale = stats + 256;
  float* gShift = stats + 384;

  gcn_prep_kernel<<<1, 128, 0, stream>>>(adj, nA, stats);
  gcn_gemm_kernel<0><<<NBT, 256, 0, stream>>>(x, nA, w, gSum, gSumSq,
                                              nullptr, nullptr, nullptr);
  gcn_finalize_kernel<<<1, 128, 0, stream>>>(gSum, gSumSq, gamma, beta,
                                             gScale, gShift);
  gcn_gemm_kernel<1><<<NBT, 256, 0, stream>>>(x, nA, w, nullptr, nullptr,
                                              gScale, gShift, out);
}
